// EncoderBlock_33784212751014
// MI455X (gfx1250) — matrix operand order repaired and checked by reading, not compiled, not run on hardware
//
#include <hip/hip_runtime.h>
#include <hip/hip_bf16.h>
#include <math.h>

typedef __attribute__((ext_vector_type(16))) __bf16 bf16x16;
typedef __attribute__((ext_vector_type(8)))  __bf16 bf16x8;
typedef __attribute__((ext_vector_type(8)))  float  f32x8;
typedef __attribute__((ext_vector_type(4)))  int    i32x4;

#define BB   16
#define SS   1024
#define DD   768
#define HH   12
#define DHD  64
#define FF   3072
#define NTOK (BB * SS)   // 16384
#define NQKV (3 * DD)    // 2304

// ---------------- async global->LDS copy (CDNA5 path) ----------------

#if defined(__has_builtin)
#  if __has_builtin(__builtin_amdgcn_global_load_async_to_lds_b128) && \
      __has_builtin(__builtin_amdgcn_s_wait_asynccnt)
#    define USE_ASYNC_LDS 1
#  endif
#endif
#ifndef USE_ASYNC_LDS
#  define USE_ASYNC_LDS 0
#endif

#if USE_ASYNC_LDS
#define AS1 __attribute__((address_space(1)))
#define AS3 __attribute__((address_space(3)))
__device__ __forceinline__ void async_b128(void* lds, const void* g) {
  // per-lane 16B copy, tracked by ASYNCcnt; no VGPR data roundtrip
  __builtin_amdgcn_global_load_async_to_lds_b128((AS1 i32x4*)g, (AS3 i32x4*)lds,
                                                 0, 0);
}
template <int N>
__device__ __forceinline__ void wait_async() {
  __builtin_amdgcn_s_wait_asynccnt(N);
}
#else
__device__ __forceinline__ void async_b128(void* lds, const void* g) {
  *(bf16x8*)lds = *(const bf16x8*)g;
}
template <int N>
__device__ __forceinline__ void wait_async() {}
#endif

// ---------------- helpers ----------------

__device__ __forceinline__ __bf16 f2bf(float f) {
  unsigned u = __builtin_bit_cast(unsigned, f);
  unsigned r = u + 0x7FFFu + ((u >> 16) & 1u);   // round-to-nearest-even
  unsigned short h = (unsigned short)(r >> 16);
  return __builtin_bit_cast(__bf16, h);
}

__device__ __forceinline__ f32x8 wmma_bf16(bf16x16 a, bf16x16 b, f32x8 c) {
  // (neg_a, A, neg_b, B, c_mod, C, reuse_a, reuse_b)
  return __builtin_amdgcn_wmma_f32_16x16x32_bf16(false, a, false, b, (short)0, c,
                                                 false, false);
}

// A fragment (16x32, bf16) from row-major LDS tile, leading dim = ld elements.
// lane<16: M=lane, K={0..7,16..23}; lane>=16: M=lane-16, K={8..15,24..31}
__device__ __forceinline__ bf16x16 load_frag_a(const __bf16* p, int ld) {
  int lane = threadIdx.x & 31;
  const __bf16* r = p + (size_t)(lane & 15) * ld + ((lane >> 4) << 3);
  bf16x16 a;
#pragma unroll
  for (int i = 0; i < 8; ++i) a[i] = r[i];
#pragma unroll
  for (int i = 0; i < 8; ++i) a[8 + i] = r[16 + i];
  return a;
}

// B fragment (32x16, bf16) from row-major [K][N] LDS tile: N = lane&15,
// lane<16: K={0..7,16..23}; lane>=16: K={8..15,24..31}.
__device__ __forceinline__ bf16x16 load_frag_b(const __bf16* p, int ld) {
  int lane = threadIdx.x & 31;
  const __bf16* r = p + (size_t)((lane >> 4) << 3) * ld + (lane & 15);
  bf16x16 b;
#pragma unroll
  for (int i = 0; i < 8; ++i) b[i] = r[(size_t)i * ld];
#pragma unroll
  for (int i = 0; i < 8; ++i) b[8 + i] = r[(size_t)(16 + i) * ld];
  return b;
}

// B fragment from a row-major [N][K] LDS tile (i.e. the transpose is needed):
// N = lane&15 row, K={0..7,16..23} (lane<16) or {8..15,24..31} (lane>=16).
__device__ __forceinline__ bf16x16 load_frag_b_tr(const __bf16* p, int ld) {
  int lane = threadIdx.x & 31;
  const __bf16* r = p + (size_t)(lane & 15) * ld + ((lane >> 4) << 3);
  bf16x16 b;
#pragma unroll
  for (int i = 0; i < 8; ++i) b[i] = r[i];
#pragma unroll
  for (int i = 0; i < 8; ++i) b[8 + i] = r[16 + i];
  return b;
}

// ---------------- weight repack ----------------

__global__ __launch_bounds__(256) void pack_wqkv_kernel(
    const float* __restrict__ wq, const float* __restrict__ wk,
    const float* __restrict__ wv, __bf16* __restrict__ out) {
  int idx = blockIdx.x * blockDim.x + threadIdx.x;
  if (idx >= DD * NQKV) return;
  int d = idx / NQKV;
  int c = idx - d * NQKV;
  int sel = c / DD;
  int cc = c - sel * DD;
  int h = cc >> 6, e = cc & 63;
  const float* src = (sel == 0) ? wq : (sel == 1) ? wk : wv;
  out[idx] = f2bf(src[((size_t)h * DD + d) * DHD + e]);  // [H,D,DH] -> [D, 3D]
}

__global__ __launch_bounds__(256) void pack_bqkv_kernel(
    const float* __restrict__ bq, const float* __restrict__ bk,
    const float* __restrict__ bv, float* __restrict__ out) {
  int c = blockIdx.x * blockDim.x + threadIdx.x;
  if (c >= NQKV) return;
  int sel = c / DD, cc = c - sel * DD;
  out[c] = (sel == 0) ? bq[cc] : (sel == 1) ? bk[cc] : bv[cc];
}

__global__ __launch_bounds__(256) void cvt_bf16_kernel(
    const float* __restrict__ in, __bf16* __restrict__ out, int n) {
  int i = blockIdx.x * blockDim.x + threadIdx.x;
  if (i < n) out[i] = f2bf(in[i]);
}

// ---------------- layernorm (fp32 -> bf16) ----------------

__global__ __launch_bounds__(256) void ln_kernel(
    const float* __restrict__ x, const float* __restrict__ g,
    const float* __restrict__ b, __bf16* __restrict__ out) {
  __shared__ float s1[256], s2[256];
  const int row = blockIdx.x;
  const int tid = threadIdx.x;
  const float* xr = x + (size_t)row * DD;
  float v0 = xr[tid], v1 = xr[tid + 256], v2 = xr[tid + 512];
  s1[tid] = v0 + v1 + v2;
  s2[tid] = v0 * v0 + v1 * v1 + v2 * v2;
  __syncthreads();
  for (int s = 128; s > 0; s >>= 1) {
    if (tid < s) { s1[tid] += s1[tid + s]; s2[tid] += s2[tid + s]; }
    __syncthreads();
  }
  float mu = s1[0] * (1.0f / DD);
  float var = s2[0] * (1.0f / DD) - mu * mu;
  float rs = rsqrtf(var + 1e-5f);
  __bf16* orow = out + (size_t)row * DD;
  orow[tid]       = f2bf((v0 - mu) * rs * g[tid]       + b[tid]);
  orow[tid + 256] = f2bf((v1 - mu) * rs * g[tid + 256] + b[tid + 256]);
  orow[tid + 512] = f2bf((v2 - mu) * rs * g[tid + 512] + b[tid + 512]);
}

// ---------------- generic WMMA GEMM ----------------
// C[M,N] = A[M,K](bf16) x W[K,N](bf16) + bias, epilogue by mode:
//   0: out_bf16 = bf16(acc+bias)
//   1: out_bf16 = bf16(gelu_erf(acc+bias))
//   2: out_f32  = acc + bias + res
// Tile 128x128x32; 256 threads = 8 waves in 2(M) x 4(N); 4x2 frags per wave.
// Double-buffered LDS with async global->LDS copies (ASYNCcnt pipeline):
// each wave issues 4 async b128 copies per tile; s_wait_asynccnt<=4 lets the
// next tile's copies stay in flight while tile t is consumed.

#define GBM 128
#define GBN 128
#define GBK 32
#define LDA (GBK + 8)    // 40 elems -> 80B row stride (16B aligned)
#define LDB (GBN + 8)    // 136 elems -> 272B row stride (16B aligned)

__global__ __launch_bounds__(256) void gemm_bf16_kernel(
    const __bf16* __restrict__ A, const __bf16* __restrict__ W,
    const float* __restrict__ bias, const float* __restrict__ res,
    float* __restrict__ outf, __bf16* __restrict__ outb,
    int M, int N, int K, int mode) {
  __shared__ __bf16 As[2][GBM * LDA];
  __shared__ __bf16 Bs[2][GBK * LDB];
  const int tid = threadIdx.x;
  const int lane = tid & 31;
  const int wid = tid >> 5;
  const int wm = wid >> 2;  // 0..1
  const int wn = wid & 3;   // 0..3
  const int bm = blockIdx.y * GBM;
  const int bn = blockIdx.x * GBN;
  const int nk = K / GBK;

  auto stage = [&](int buf, int k0) {
    // A tile 128x32: 2 b128 per thread
#pragma unroll
    for (int v = tid; v < 512; v += 256) {
      int r = v >> 2, c = (v & 3) << 3;
      async_b128(&As[buf][r * LDA + c], A + (size_t)(bm + r) * K + k0 + c);
    }
    // B tile 32x128: 2 b128 per thread
#pragma unroll
    for (int v = tid; v < 512; v += 256) {
      int r = v >> 4, c = (v & 15) << 3;
      async_b128(&Bs[buf][r * LDB + c], W + (size_t)(k0 + r) * N + bn + c);
    }
  };

  f32x8 acc[4][2];
#pragma unroll
  for (int i = 0; i < 4; ++i)
#pragma unroll
    for (int j = 0; j < 2; ++j)
      acc[i][j] = f32x8{0.f, 0.f, 0.f, 0.f, 0.f, 0.f, 0.f, 0.f};

  stage(0, 0);
  for (int t = 0; t < nk; ++t) {
    const int cur = t & 1;
    if (t + 1 < nk) {
      stage(cur ^ 1, (t + 1) * GBK);
      wait_async<4>();   // tile t's (older) copies done; t+1's may be in flight
    } else {
      wait_async<0>();
    }
    __syncthreads();

    bf16x16 af[4];
    bf16x16 bfm[2];
#pragma unroll
    for (int i = 0; i < 4; ++i)
      af[i] = load_frag_a(&As[cur][(wm * 64 + i * 16) * LDA], LDA);
#pragma unroll
    for (int j = 0; j < 2; ++j)
      bfm[j] = load_frag_b(&Bs[cur][wn * 32 + j * 16], LDB);
#pragma unroll
    for (int i = 0; i < 4; ++i)
#pragma unroll
      for (int j = 0; j < 2; ++j)
        acc[i][j] = wmma_bf16(af[i], bfm[j], acc[i][j]);
    __syncthreads();   // everyone done reading before buffer reuse at t+2
  }

  const int ch = lane >> 4, cn = lane & 15;
#pragma unroll
  for (int i = 0; i < 4; ++i) {
#pragma unroll
    for (int j = 0; j < 2; ++j) {
      int col = bn + wn * 32 + j * 16 + cn;
      float bcol = bias[col];
#pragma unroll
      for (int r = 0; r < 8; ++r) {
        int row = bm + wm * 64 + i * 16 + r + 8 * ch;
        size_t off = (size_t)row * N + col;
        float v = acc[i][j][r] + bcol;
        if (mode == 0) {
          outb[off] = f2bf(v);
        } else if (mode == 1) {
          float gl = 0.5f * v * (1.0f + erff(v * 0.70710678118654752f));
          outb[off] = f2bf(gl);
        } else {
          outf[off] = v + res[off];
        }
      }
    }
  }
}

// ---------------- flash attention ----------------
// grid: (S/64 q-tiles, H, B); 128 threads = 4 waves, each wave owns 16 Q rows.
// Q/K/V tiles staged with async copies; K stays row-major and the score
// B-fragment is a strided (conflict-free) LDS gather of K^T.

#define APAD 8
#define ALD (64 + APAD)  // 72 elems -> 144B row stride (16B aligned)

__global__ __launch_bounds__(128) void attn_kernel(
    const __bf16* __restrict__ qkv, __bf16* __restrict__ o) {
  __shared__ __bf16 Qs[64 * ALD];   // [q][d]
  __shared__ __bf16 Ks[64 * ALD];   // [key][d] (row-major; transposed on read)
  __shared__ __bf16 Vs[64 * ALD];   // [key][d]
  __shared__ __bf16 Ps[4][16 * ALD];

  const int tid = threadIdx.x;
  const int lane = tid & 31;
  const int w = tid >> 5;
  const int qt = blockIdx.x;
  const int h = blockIdx.y;
  const int b = blockIdx.z;
  const size_t rowbase = (size_t)(b * SS + qt * 64);

  // load Q tile (64 x 64): 4 async b128 per thread
#pragma unroll
  for (int v = tid; v < 512; v += 128) {
    int r = v >> 3, c = (v & 7) << 3;
    async_b128(&Qs[r * ALD + c],
               qkv + (rowbase + r) * NQKV + h * DHD + c);
  }
  wait_async<0>();
  __syncthreads();

  float run_m[8], run_l[8];
  f32x8 oacc[4];
#pragma unroll
  for (int r = 0; r < 8; ++r) { run_m[r] = -1e30f; run_l[r] = 0.f; }
#pragma unroll
  for (int j = 0; j < 4; ++j) oacc[j] = f32x8{0.f,0.f,0.f,0.f,0.f,0.f,0.f,0.f};

  for (int kt = 0; kt < SS / 64; ++kt) {
    __syncthreads();  // previous iteration's readers done before overwrite
    const size_t kbase = (size_t)(b * SS + kt * 64);
    // K and V tiles row-major: 4+4 async b128 per thread
#pragma unroll
    for (int v = tid; v < 512; v += 128) {
      int r = v >> 3, c = (v & 7) << 3;
      async_b128(&Ks[r * ALD + c],
                 qkv + (kbase + r) * NQKV + DD + h * DHD + c);
      async_b128(&Vs[r * ALD + c],
                 qkv + (kbase + r) * NQKV + 2 * DD + h * DHD + c);
    }
    wait_async<0>();
    __syncthreads();

    // scores: wave's 16 x 64 strip = Q(16x64) x K^T(64x64)
    f32x8 s[4];
#pragma unroll
    for (int j = 0; j < 4; ++j) {
      s[j] = f32x8{0.f, 0.f, 0.f, 0.f, 0.f, 0.f, 0.f, 0.f};
#pragma unroll
      for (int kk = 0; kk < 2; ++kk) {
        bf16x16 a = load_frag_a(&Qs[(w * 16) * ALD + kk * 32], ALD);
        bf16x16 bb = load_frag_b_tr(&Ks[(j * 16) * ALD + kk * 32], ALD);
        s[j] = wmma_bf16(a, bb, s[j]);
      }
    }
    // online softmax; row r lives in lanes sharing (lane>>4), cols across lane&15
#pragma unroll
    for (int r = 0; r < 8; ++r) {
      float tm = -1e30f;
#pragma unroll
      for (int j = 0; j < 4; ++j) {
        s[j][r] *= 0.125f;  // DH^-0.5
        tm = fmaxf(tm, s[j][r]);
      }
      tm = fmaxf(tm, __shfl_xor(tm, 1, 32));
      tm = fmaxf(tm, __shfl_xor(tm, 2, 32));
      tm = fmaxf(tm, __shfl_xor(tm, 4, 32));
      tm = fmaxf(tm, __shfl_xor(tm, 8, 32));
      float nm = fmaxf(run_m[r], tm);
      float corr = __expf(run_m[r] - nm);
      run_m[r] = nm;
      float ts = 0.f;
#pragma unroll
      for (int j = 0; j < 4; ++j) {
        float p = __expf(s[j][r] - nm);
        s[j][r] = p;
        ts += p;
      }
      ts += __shfl_xor(ts, 1, 32);
      ts += __shfl_xor(ts, 2, 32);
      ts += __shfl_xor(ts, 4, 32);
      ts += __shfl_xor(ts, 8, 32);
      run_l[r] = run_l[r] * corr + ts;
#pragma unroll
      for (int j = 0; j < 4; ++j) oacc[j][r] *= corr;
    }
    // spill P (D-layout) to wave-private LDS, reload as A fragments
    {
      int mrow = (lane >> 4) * 8, ncol = lane & 15;
#pragma unroll
      for (int j = 0; j < 4; ++j)
#pragma unroll
        for (int r = 0; r < 8; ++r)
          Ps[w][(mrow + r) * ALD + j * 16 + ncol] = f2bf(s[j][r]);
    }
    // O += P(16x64) x V(64x64)
#pragma unroll
    for (int jo = 0; jo < 4; ++jo) {
#pragma unroll
      for (int kk = 0; kk < 2; ++kk) {
        bf16x16 a = load_frag_a(&Ps[w][kk * 32], ALD);
        bf16x16 bb = load_frag_b(&Vs[(kk * 32) * ALD + jo * 16], ALD);
        oacc[jo] = wmma_bf16(a, bb, oacc[jo]);
      }
    }
  }

  // normalize and store concat-head layout [B,S,H*DH] as bf16
  {
    int mrow = (lane >> 4) * 8, ncol = lane & 15;
#pragma unroll
    for (int r = 0; r < 8; ++r) {
      float inv = (run_l[r] > 0.f) ? 1.f / run_l[r] : 0.f;
      size_t srow = rowbase + w * 16 + mrow + r;
#pragma unroll
      for (int jo = 0; jo < 4; ++jo)
        o[srow * DD + h * DHD + jo * 16 + ncol] = f2bf(oacc[jo][r] * inv);
    }
  }
}

// ---------------- host orchestration ----------------

extern "C" void kernel_launch(void* const* d_in, const int* in_sizes, int n_in,
                              void* d_out, int out_size, void* d_ws,
                              size_t ws_size, hipStream_t stream) {
  (void)in_sizes; (void)n_in; (void)out_size;
  const float* x     = (const float*)d_in[0];
  const float* ln1_g = (const float*)d_in[1];
  const float* ln1_b = (const float*)d_in[2];
  const float* wq    = (const float*)d_in[3];
  const float* bq    = (const float*)d_in[4];
  const float* wk    = (const float*)d_in[5];
  const float* bk    = (const float*)d_in[6];
  const float* wv    = (const float*)d_in[7];
  const float* bv    = (const float*)d_in[8];
  const float* wo    = (const float*)d_in[9];
  const float* bo    = (const float*)d_in[10];
  const float* ln2_g = (const float*)d_in[11];
  const float* ln2_b = (const float*)d_in[12];
  const float* w1    = (const float*)d_in[13];
  const float* b1    = (const float*)d_in[14];
  const float* w2    = (const float*)d_in[15];
  const float* b2    = (const float*)d_in[16];
  float* out = (float*)d_out;

  char* ws = (char*)d_ws;
  size_t off = 0;
  auto alloc = [&](size_t bytes) -> char* {
    char* p = ws + off;
    off += (bytes + 255) & ~(size_t)255;
    return p;
  };
  __bf16* wqkv_bf = (__bf16*)alloc((size_t)DD * NQKV * 2);
  __bf16* wo_bf   = (__bf16*)alloc((size_t)DD * DD * 2);
  __bf16* w1_bf   = (__bf16*)alloc((size_t)DD * FF * 2);
  __bf16* w2_bf   = (__bf16*)alloc((size_t)FF * DD * 2);
  float*  bqkv_f  = (float*)alloc((size_t)NQKV * 4);
  __bf16* h_bf    = (__bf16*)alloc((size_t)NTOK * DD * 2);
  __bf16* qkv_bf  = (__bf16*)alloc((size_t)NTOK * NQKV * 2);
  __bf16* att_bf  = (__bf16*)alloc((size_t)NTOK * DD * 2);
  float*  x2_f    = (float*)alloc((size_t)NTOK * DD * 4);
  // qkv_bf + att_bf regions are contiguous (each size is 256B-aligned) and
  // together are exactly NTOK*FF bf16 -> reuse for the MLP hidden activations.
  __bf16* m_bf = qkv_bf;
  if (ws_size < off) return;

  // weight repack (tiny vs. GEMM cost)
  pack_wqkv_kernel<<<(DD * NQKV + 255) / 256, 256, 0, stream>>>(wq, wk, wv, wqkv_bf);
  pack_bqkv_kernel<<<(NQKV + 255) / 256, 256, 0, stream>>>(bq, bk, bv, bqkv_f);
  cvt_bf16_kernel<<<(DD * DD + 255) / 256, 256, 0, stream>>>(wo, wo_bf, DD * DD);
  cvt_bf16_kernel<<<(DD * FF + 255) / 256, 256, 0, stream>>>(w1, w1_bf, DD * FF);
  cvt_bf16_kernel<<<(FF * DD + 255) / 256, 256, 0, stream>>>(w2, w2_bf, FF * DD);

  // ln1 -> fused QKV GEMM -> flash attention -> O proj (+residual)
  ln_kernel<<<NTOK, 256, 0, stream>>>(x, ln1_g, ln1_b, h_bf);
  gemm_bf16_kernel<<<dim3(NQKV / GBN, NTOK / GBM), 256, 0, stream>>>(
      h_bf, wqkv_bf, bqkv_f, nullptr, nullptr, qkv_bf, NTOK, NQKV, DD, 0);
  attn_kernel<<<dim3(SS / 64, HH, BB), 128, 0, stream>>>(qkv_bf, att_bf);
  gemm_bf16_kernel<<<dim3(DD / GBN, NTOK / GBM), 256, 0, stream>>>(
      att_bf, wo_bf, bo, x, x2_f, nullptr, NTOK, DD, DD, 2);

  // ln2 -> MLP1 (GELU) -> MLP2 (+residual) -> out
  ln_kernel<<<NTOK, 256, 0, stream>>>(x2_f, ln2_g, ln2_b, h_bf);
  gemm_bf16_kernel<<<dim3(FF / GBN, NTOK / GBM), 256, 0, stream>>>(
      h_bf, w1_bf, b1, nullptr, nullptr, m_bf, NTOK, FF, DD, 1);
  gemm_bf16_kernel<<<dim3(DD / GBN, NTOK / GBM), 256, 0, stream>>>(
      m_bf, w2_bf, b2, x2_f, out, nullptr, NTOK, DD, FF, 2);
}
